// Model_12919261626633
// MI455X (gfx1250) — compile-verified, matrix operand order checked by reading, not run on hardware
//
#include <hip/hip_runtime.h>
#include <math.h>

// ---------------- types ----------------
typedef __attribute__((ext_vector_type(16))) __bf16 v16bf;
typedef __attribute__((ext_vector_type(8)))  float  v8f;
typedef unsigned short u16;
typedef unsigned int   u32;

#define B_ROWS   1024
#define QCAND    32768
#define D_IN     256
#define D_MAIN   512
#define D_BLOCK  1024
#define CTX      96
#define NCLS     10

// round-to-nearest-even f32 -> bf16 (raw u16)
static __device__ __forceinline__ u16 f2bf(float f) {
  u32 u = __builtin_bit_cast(u32, f);
  u32 r = u + 0x7FFFu + ((u >> 16) & 1u);
  return (u16)(r >> 16);
}

// monotone map f32 -> u32 for radix select (ascending)
static __device__ __forceinline__ u32 mono(float f) {
  u32 u = __builtin_bit_cast(u32, f);
  return (u & 0x80000000u) ? ~u : (u | 0x80000000u);
}

// LDS byte offset of a __shared__ pointer (low 32 bits of the flat address)
static __device__ __forceinline__ unsigned lds_off(const void* p) {
  return (unsigned)(reinterpret_cast<uintptr_t>(p));
}

// async 16B global -> LDS copy (per-lane), tracked with ASYNCcnt
static __device__ __forceinline__ void async_b128(unsigned lds, const void* g) {
  asm volatile("global_load_async_to_lds_b128 %0, %1, off"
               :: "v"(lds), "v"(g) : "memory");
}
template<int N>
static __device__ __forceinline__ void wait_async() {
  asm volatile("s_wait_asynccnt %0" :: "i"(N) : "memory");
}

// Build a 16-element bf16 fragment from a 32-element (64B) LDS row.
// CDNA5 16-bit A/B layout: lane<16 -> K = {0..7,16..23}, lane>=16 -> +8.
static __device__ __forceinline__ v16bf load_frag(const u16* rowp, int khalf) {
  const u32* p = (const u32*)rowp;
  union { v16bf v; u32 u[8]; } f;
  const int b = khalf * 4;
#pragma unroll
  for (int i = 0; i < 4; ++i) {
    f.u[i]     = p[b + i];
    f.u[4 + i] = p[8 + b + i];
  }
  return f.v;
}

// ---------------- bf16 WMMA GEMM: C = epilog(A[M,K] @ Bt[N,K]^T) ----------------
// Block tile 256x64, 8 waves, wave tile 32x64 (2 A-frags x 4 B-frags = 8 WMMA).
// Double-buffered LDS, async global->LDS staging (5x b128 per lane per tile).
// EPILOG 0: +bias(+relu)  1: c2[n]-2*acc  2: +bias+resid.  WB16: bf16 copy out.
template<int RELU, int EPILOG, int WB16>
__global__ __launch_bounds__(256)
void wmma_gemm(const u16* __restrict__ A, const u16* __restrict__ Bt,
               const float* __restrict__ bias, const float* __restrict__ resid,
               const float* __restrict__ c2v, float* __restrict__ C,
               u16* __restrict__ Cb, int M, int N, int K, int m_base)
{
  (void)M;
  __shared__ u16 sA[2][256 * 32];   // [m][k] 16 KB per buffer
  __shared__ u16 sB[2][64 * 32];    // [n][k]  4 KB per buffer
  const int tid  = threadIdx.x;
  const int lane = tid & 31;
  const int wv   = tid >> 5;
  const int n0   = blockIdx.x * 64;
  const int m0   = blockIdx.y * 256;
  const int mA   = m_base + m0;

  v8f zero = {};
  v8f acc[2][4];
#pragma unroll
  for (int s = 0; s < 2; ++s)
#pragma unroll
    for (int i = 0; i < 4; ++i) acc[s][i] = zero;

  // async staging plan per tile: A 4x16B, B 1x16B per lane
  const int cA = (tid & 3) * 8;                 // k offset (elems)
  const int rA = tid >> 2;                      // + 64*t
  const u16* gA[4];
  unsigned   lA0[4], lA1[4];
#pragma unroll
  for (int t = 0; t < 4; ++t) {
    int r = rA + 64 * t;
    gA[t]  = A + (size_t)(mA + r) * K + cA;
    lA0[t] = lds_off(&sA[0][r * 32 + cA]);
    lA1[t] = lds_off(&sA[1][r * 32 + cA]);
  }
  const u16* gB = Bt + (size_t)(n0 + rA) * K + cA;
  const unsigned lB0 = lds_off(&sB[0][rA * 32 + cA]);
  const unsigned lB1 = lds_off(&sB[1][rA * 32 + cA]);

  const int frag_idx = lane & 15;
  const int khalf    = lane >> 4;
  const int nsteps   = K >> 5;

  // prologue: stage tile 0 into buffer 0
#pragma unroll
  for (int t = 0; t < 4; ++t) async_b128(lA0[t], gA[t]);
  async_b128(lB0, gB);

  for (int i = 0; i < nsteps; ++i) {
    if (i + 1 < nsteps) {
      const int k1 = (i + 1) * 32;
      if ((i + 1) & 1) {
#pragma unroll
        for (int t = 0; t < 4; ++t) async_b128(lA1[t], gA[t] + k1);
        async_b128(lB1, gB + k1);
      } else {
#pragma unroll
        for (int t = 0; t < 4; ++t) async_b128(lA0[t], gA[t] + k1);
        async_b128(lB0, gB + k1);
      }
      wait_async<5>();     // tile i complete (in-order), tile i+1 in flight
    } else {
      wait_async<0>();
    }
    __syncthreads();

    const u16* bufA = &sA[i & 1][0];
    const u16* bufB = &sB[i & 1][0];
    v16bf a0 = load_frag(&bufA[(wv * 32 +      frag_idx) * 32], khalf);
    v16bf a1 = load_frag(&bufA[(wv * 32 + 16 + frag_idx) * 32], khalf);
#pragma unroll
    for (int nb = 0; nb < 4; ++nb) {
      v16bf b = load_frag(&bufB[(nb * 16 + frag_idx) * 32], khalf);
      acc[0][nb] = __builtin_amdgcn_wmma_f32_16x16x32_bf16(
          false, a0, false, b, (short)0, acc[0][nb], false, false);
      acc[1][nb] = __builtin_amdgcn_wmma_f32_16x16x32_bf16(
          false, a1, false, b, (short)0, acc[1][nb], false, false);
    }
    __syncthreads();
  }

  const int ncol = lane & 15;
  const int mrow = (lane >> 4) * 8;
#pragma unroll
  for (int s = 0; s < 2; ++s) {
#pragma unroll
    for (int nb = 0; nb < 4; ++nb) {
      int col = n0 + nb * 16 + ncol;
      if (col >= N) continue;
#pragma unroll
      for (int j = 0; j < 8; ++j) {
        int row = m0 + wv * 32 + s * 16 + mrow + j;
        float v = acc[s][nb][j];
        if (EPILOG == 1) {
          v = c2v[col] - 2.0f * v;        // L2 ranking key (q2 is row-const)
        } else {
          if (bias) v += bias[col];
          if (RELU) v = v > 0.0f ? v : 0.0f;
          if (EPILOG == 2) v += resid[(size_t)row * N + col];
        }
        C[(size_t)row * N + col] = v;
        if (WB16) Cb[(size_t)row * N + col] = f2bf(v);
      }
    }
  }
}

// ---------------- f32 -> bf16 bulk convert (8 elems/thread) ----------------
__global__ __launch_bounds__(256)
void convert_bf16_kernel(const float* __restrict__ in, u16* __restrict__ out,
                         long n)
{
  long i = ((long)blockIdx.x * 256 + threadIdx.x) * 8;
  if (i + 8 > n) return;
  const float4* p = (const float4*)(in + i);
  float4 a = p[0], b = p[1];
  union { u16 h[8]; uint4 v; } o;
  o.h[0] = f2bf(a.x); o.h[1] = f2bf(a.y); o.h[2] = f2bf(a.z); o.h[3] = f2bf(a.w);
  o.h[4] = f2bf(b.x); o.h[5] = f2bf(b.y); o.h[6] = f2bf(b.z); o.h[7] = f2bf(b.w);
  *(uint4*)(out + i) = o.v;
}

// ---------------- W[K,N] f32 -> Wt[NP,K] bf16 (zero-padded rows) ----------------
__global__ __launch_bounds__(256)
void transpose_bf16_kernel(const float* __restrict__ W, u16* __restrict__ Wt,
                           int K, int N)
{
  int n = blockIdx.y;
  int k = blockIdx.x * 256 + threadIdx.x;
  if (k >= K) return;
  float v = (n < N) ? W[(size_t)k * N + n] : 0.0f;
  Wt[(size_t)n * K + k] = f2bf(v);
}

// ---------------- row squared-norms (wave per row, float4) ----------------
__global__ __launch_bounds__(256)
void rownorm_kernel(const float* __restrict__ X, float* __restrict__ out,
                    int nrows, int K)
{
  int wv = threadIdx.x >> 5, lane = threadIdx.x & 31;
  int row = blockIdx.x * 8 + wv;
  if (row >= nrows) return;
  const float4* p = (const float4*)(X + (size_t)row * K);
  float s = 0.f;
  for (int k = lane; k < (K >> 2); k += 32) {
    float4 v = p[k];
    s += v.x * v.x + v.y * v.y + v.z * v.z + v.w * v.w;
  }
#pragma unroll
  for (int off = 16; off > 0; off >>= 1) s += __shfl_xor(s, off, 32);
  if (lane == 0) out[row] = s;
}

// ---------------- per-row top-96 smallest via 4-pass radix select ----------------
__global__ __launch_bounds__(256)
void topk_kernel(const float* __restrict__ scores, int* __restrict__ idx_out,
                 int row_base, int Q, int KSEL)
{
  const float* vals = scores + (size_t)blockIdx.x * Q;
  int* outp = idx_out + (size_t)(row_base + blockIdx.x) * KSEL;
  __shared__ u32 hist[256];
  __shared__ u32 sh_prefix;
  __shared__ int sh_rem;
  __shared__ u32 sh_less, sh_tie;
  const int tid = threadIdx.x;

  if (tid == 0) { sh_prefix = 0u; sh_rem = KSEL; }
  __syncthreads();

  for (int pass = 0; pass < 4; ++pass) {
    int shift = 24 - pass * 8;
    hist[tid] = 0u;
    __syncthreads();
    u32 prefix = sh_prefix;
    for (int n = tid; n < Q; n += 256) {
      u32 u = mono(vals[n]);
      if (pass == 0 || (u >> (shift + 8)) == prefix)
        atomicAdd(&hist[(u >> shift) & 255u], 1u);
    }
    __syncthreads();
    if (tid == 0) {
      int rem = sh_rem;
      u32 cum = 0; int v = 0;
      for (; v < 255; ++v) {
        u32 h = hist[v];
        if (cum + h >= (u32)rem) break;
        cum += h;
      }
      sh_prefix = (prefix << 8) | (u32)v;
      sh_rem = rem - (int)cum;
    }
    __syncthreads();
  }

  if (tid == 0) { sh_less = 0u; sh_tie = 0u; }
  __syncthreads();
  const u32 T = sh_prefix;
  const int ties = sh_rem;
  const int base_less = KSEL - ties;
  for (int n = tid; n < Q; n += 256) {
    u32 u = mono(vals[n]);
    if (u < T) {
      u32 p = atomicAdd(&sh_less, 1u);
      outp[p] = n;
    } else if (u == T) {
      u32 t = atomicAdd(&sh_tie, 1u);
      if ((int)t < ties) outp[base_less + (int)t] = n;
    }
  }
}

// ---------------- exact f32 distance softmax + context + residual ----------------
__global__ __launch_bounds__(256)
void context_kernel(const float* __restrict__ query, const float* __restrict__ ck,
                    const float* __restrict__ cv, const int* __restrict__ idx,
                    float* __restrict__ xout, u16* __restrict__ xoutb,
                    int D, int KSEL)
{
  const int row = blockIdx.x;
  __shared__ float qs[D_MAIN];
  __shared__ int   is[CTX];
  __shared__ float cross_s[CTX], kk_s[CTX], probs[CTX];
  __shared__ float red[8];
  const int tid = threadIdx.x, lane = tid & 31, wv = tid >> 5;

  for (int d = tid; d < D; d += 256) qs[d] = query[(size_t)row * D + d];
  if (tid < KSEL) is[tid] = idx[(size_t)row * KSEL + tid];
  __syncthreads();

  float part = 0.f;
  for (int d = tid; d < D; d += 256) { float v = qs[d]; part += v * v; }
#pragma unroll
  for (int off = 16; off > 0; off >>= 1) part += __shfl_xor(part, off, 32);
  if (lane == 0) red[wv] = part;
  __syncthreads();
  if (tid == 0) { float s = 0.f; for (int i = 0; i < 8; ++i) s += red[i]; red[0] = s; }
  __syncthreads();

  for (int j = wv; j < KSEL; j += 8) {
    const float4* kp = (const float4*)(ck + (size_t)is[j] * D);
    float c = 0.f, kk = 0.f;
    for (int d4 = lane; d4 < (D >> 2); d4 += 32) {
      float4 kv = kp[d4];
      int d = d4 * 4;
      c  += qs[d] * kv.x + qs[d+1] * kv.y + qs[d+2] * kv.z + qs[d+3] * kv.w;
      kk += kv.x * kv.x + kv.y * kv.y + kv.z * kv.z + kv.w * kv.w;
    }
#pragma unroll
    for (int off = 16; off > 0; off >>= 1) {
      c  += __shfl_xor(c, off, 32);
      kk += __shfl_xor(kk, off, 32);
    }
    if (lane == 0) { cross_s[j] = c; kk_s[j] = kk; }
  }
  __syncthreads();

  if (tid == 0) {
    float q2 = red[0];
    float dmin = 3.4e38f;
    for (int j = 0; j < KSEL; ++j) {
      float d2 = q2 + kk_s[j] - 2.f * cross_s[j];
      float dist = sqrtf(d2 > 0.f ? d2 : 0.f);   // TEMP == 1.0
      probs[j] = dist;
      dmin = dist < dmin ? dist : dmin;
    }
    float ssum = 0.f;
    for (int j = 0; j < KSEL; ++j) { float e = expf(dmin - probs[j]); probs[j] = e; ssum += e; }
    float inv = 1.f / ssum;
    for (int j = 0; j < KSEL; ++j) probs[j] *= inv;
  }
  __syncthreads();

  for (int d = tid; d < D; d += 256) {
    float acc = qs[d];
    for (int j = 0; j < KSEL; ++j) acc += probs[j] * cv[(size_t)is[j] * D + d];
    xout[(size_t)row * D + d]  = acc;
    xoutb[(size_t)row * D + d] = f2bf(acc);
  }
}

// ---------------- launch ----------------
#define MB (size_t)(1024 * 1024)
#define KB (size_t)1024

extern "C" void kernel_launch(void* const* d_in, const int* in_sizes, int n_in,
                              void* d_out, int out_size, void* d_ws, size_t ws_size,
                              hipStream_t stream)
{
  (void)in_sizes; (void)n_in; (void)out_size; (void)ws_size;
  // params flattened as JAX pytree (dict keys sorted): enc, head, pred, w_in
  const float* x_num = (const float*)d_in[0];
  const float* candk = (const float*)d_in[1];
  const float* candv = (const float*)d_in[2];
  const float* e0W1 = (const float*)d_in[4];
  const float* e0b1 = (const float*)d_in[5];
  const float* e0W2 = (const float*)d_in[6];
  const float* e0b2 = (const float*)d_in[7];
  const float* e1W1 = (const float*)d_in[8];
  const float* e1b1 = (const float*)d_in[9];
  const float* e1W2 = (const float*)d_in[10];
  const float* e1b2 = (const float*)d_in[11];
  const float* hW   = (const float*)d_in[12];
  const float* hbias= (const float*)d_in[13];
  const float* pW1  = (const float*)d_in[14];
  const float* pb1  = (const float*)d_in[15];
  const float* pW2  = (const float*)d_in[16];
  const float* pb2  = (const float*)d_in[17];
  const float* wiW  = (const float*)d_in[18];
  const float* wib  = (const float*)d_in[19];
  float* out = (float*)d_out;

  char* ws = (char*)d_ws;
  float* x0   = (float*)(ws);                       // 1024x512 f32
  float* x1   = (float*)(ws + 2 * MB);              // 1024x512 f32
  float* hbuf = (float*)(ws + 4 * MB);              // 1024x1024 f32
  float* c2v  = (float*)(ws + 8 * MB);              // 32768 f32
  int*   idxb = (int*)  (ws + 8 * MB + 128 * KB);   // 1024x96
  u16* x0b  = (u16*)(ws + 8 * MB + 512 * KB);       // 1 MB
  u16* x1b  = (u16*)(ws + 9 * MB + 512 * KB);       // 1 MB
  u16* hbufb= (u16*)(ws + 10 * MB + 512 * KB);      // 2 MB
  u16* xnb  = (u16*)(ws + 12 * MB + 512 * KB);      // 512 KB
  u16* wiT  = (u16*)(ws + 13 * MB);                 // 256 KB
  u16* e0W1T= (u16*)(ws + 13 * MB + 256 * KB);      // 1 MB
  u16* e0W2T= (u16*)(ws + 14 * MB + 256 * KB);      // 1 MB
  u16* e1W1T= (u16*)(ws + 15 * MB + 256 * KB);      // 1 MB
  u16* e1W2T= (u16*)(ws + 16 * MB + 256 * KB);      // 1 MB
  u16* pW1T = (u16*)(ws + 17 * MB + 256 * KB);      // 512 KB
  u16* pW2T = (u16*)(ws + 17 * MB + 768 * KB);      // 512 KB
  u16* hWT  = (u16*)(ws + 18 * MB + 256 * KB);      // 64 KB (padded)
  u16* ckb  = (u16*)(ws + 19 * MB);                 // 32 MB
  float* d2p= (float*)(ws + 51 * MB);               // 256x32768 f32 (32 MB)

  const dim3 blk(256);

  // ---- precision prep: bf16 copies (candidates stay L2-resident: 32 MB) ----
  convert_bf16_kernel<<<(B_ROWS * D_IN) / 2048, blk, 0, stream>>>(x_num, xnb, (long)B_ROWS * D_IN);
  convert_bf16_kernel<<<((long)QCAND * D_MAIN) / 2048, blk, 0, stream>>>(candk, ckb, (long)QCAND * D_MAIN);
  transpose_bf16_kernel<<<dim3(1, D_MAIN),  blk, 0, stream>>>(wiW,  wiT,   D_IN,   D_MAIN);
  transpose_bf16_kernel<<<dim3(2, D_BLOCK), blk, 0, stream>>>(e0W1, e0W1T, D_MAIN, D_BLOCK);
  transpose_bf16_kernel<<<dim3(4, D_MAIN),  blk, 0, stream>>>(e0W2, e0W2T, D_BLOCK, D_MAIN);
  transpose_bf16_kernel<<<dim3(2, D_BLOCK), blk, 0, stream>>>(e1W1, e1W1T, D_MAIN, D_BLOCK);
  transpose_bf16_kernel<<<dim3(4, D_MAIN),  blk, 0, stream>>>(e1W2, e1W2T, D_BLOCK, D_MAIN);
  transpose_bf16_kernel<<<dim3(2, D_MAIN),  blk, 0, stream>>>(pW1,  pW1T,  D_MAIN, D_MAIN);
  transpose_bf16_kernel<<<dim3(2, D_MAIN),  blk, 0, stream>>>(pW2,  pW2T,  D_MAIN, D_MAIN);
  transpose_bf16_kernel<<<dim3(2, 64),      blk, 0, stream>>>(hW,   hWT,   D_MAIN, NCLS);

  rownorm_kernel<<<QCAND / 8, blk, 0, stream>>>(candk, c2v, QCAND, D_MAIN);

  // ---- encoder ----
  wmma_gemm<0,0,1><<<dim3(D_MAIN/64,  B_ROWS/256), blk, 0, stream>>>(
      xnb,   wiT,   wib,  nullptr, nullptr, x0,   x0b,   B_ROWS, D_MAIN,  D_IN,   0);
  wmma_gemm<1,0,1><<<dim3(D_BLOCK/64, B_ROWS/256), blk, 0, stream>>>(
      x0b,   e0W1T, e0b1, nullptr, nullptr, hbuf, hbufb, B_ROWS, D_BLOCK, D_MAIN, 0);
  wmma_gemm<0,0,1><<<dim3(D_MAIN/64,  B_ROWS/256), blk, 0, stream>>>(
      hbufb, e0W2T, e0b2, nullptr, nullptr, x1,   x1b,   B_ROWS, D_MAIN,  D_BLOCK,0);
  wmma_gemm<1,0,1><<<dim3(D_BLOCK/64, B_ROWS/256), blk, 0, stream>>>(
      x1b,   e1W1T, e1b1, nullptr, nullptr, hbuf, hbufb, B_ROWS, D_BLOCK, D_MAIN, 0);
  wmma_gemm<0,0,1><<<dim3(D_MAIN/64,  B_ROWS/256), blk, 0, stream>>>(
      hbufb, e1W2T, e1b2, nullptr, nullptr, x0,   x0b,   B_ROWS, D_MAIN,  D_BLOCK,0);
  // query: f32 in x0, bf16 in x0b

  // ---- retrieval + top-96, chunked by 256 query rows ----
  for (int chunk = 0; chunk < B_ROWS / 256; ++chunk) {
    int m_base = chunk * 256;
    wmma_gemm<0,1,0><<<dim3(QCAND/64, 1), blk, 0, stream>>>(
        x0b, ckb, nullptr, nullptr, c2v, d2p, nullptr, B_ROWS, QCAND, D_MAIN, m_base);
    topk_kernel<<<256, blk, 0, stream>>>(d2p, idxb, m_base, QCAND, CTX);
  }

  // ---- exact-f32 softmax context + residual -> x (x1 / x1b) ----
  context_kernel<<<B_ROWS, blk, 0, stream>>>(x0, candk, candv, idxb,
                                             x1, x1b, D_MAIN, CTX);

  // ---- predictor residual block ----
  wmma_gemm<1,0,1><<<dim3(D_MAIN/64, B_ROWS/256), blk, 0, stream>>>(
      x1b,   pW1T, pb1, nullptr, nullptr, hbuf, hbufb, B_ROWS, D_MAIN, D_MAIN, 0);
  wmma_gemm<0,2,1><<<dim3(D_MAIN/64, B_ROWS/256), blk, 0, stream>>>(
      hbufb, pW2T, pb2, x1,      nullptr, x0,   x0b,   B_ROWS, D_MAIN, D_MAIN, 0);

  // ---- head (N=10, weight rows zero-padded to 64) ----
  wmma_gemm<0,0,0><<<dim3(1, B_ROWS/256), blk, 0, stream>>>(
      x0b, hWT, hbias, nullptr, nullptr, out, nullptr, B_ROWS, NCLS, D_MAIN, 0);
}
